// CoSFuserMoELayer_11836929868010
// MI455X (gfx1250) — matrix operand order repaired and checked by reading, not compiled, not run on hardware
//
#include <hip/hip_runtime.h>
#include <hip/hip_bf16.h>
#include <math.h>

// ---------------- problem constants ----------------
#define T_TOK 8192      // B*S tokens
#define Dm    1024
#define Hm    4096
#define Em    8
#define BMt   64        // row tile (tokens)
#define BNt   128       // col tile
#define BKt   32        // k step (matches wmma K for bf16)
#define ROW_TILES (T_TOK / BMt + Em)   // 136 (worst-case padded tiles)
#define PADT  (ROW_TILES * BMt)        // 8704 padded rows
#define LDA   40        // padded LDS row strides (bank-conflict relief)
#define LDB   40

typedef __bf16 v16bf __attribute__((ext_vector_type(16)));
typedef float  v8f   __attribute__((ext_vector_type(8)));

union Frag { v16bf v; uint4 q[2]; };

__device__ __forceinline__ unsigned short f2bf(float f) {
    unsigned int u = __float_as_uint(f);
    // round-to-nearest-even bf16
    return (unsigned short)((u + 0x7FFFu + ((u >> 16) & 1u)) >> 16);
}

__device__ __forceinline__ float gelu_tanh(float v) {
    const float k0 = 0.7978845608028654f;   // sqrt(2/pi)
    float inner = k0 * (v + 0.044715f * v * v * v);
    return 0.5f * v * (1.0f + tanhf(inner));
}

// LDS byte address of a generic pointer into __shared__ (ISA 10.2: LDS aperture
// maps to LDS by truncating to addr[31:0]).
__device__ __forceinline__ unsigned lds_addr(const void* p) {
    return (unsigned)(unsigned long long)p;
}

// Direct async global -> LDS 16B copy (gfx1250, tracked by ASYNCcnt).
__device__ __forceinline__ void async_copy_b128(unsigned lds_off, const void* gptr) {
    asm volatile("global_load_async_to_lds_b128 %0, %1, off"
                 :: "v"(lds_off), "v"((unsigned long long)(size_t)gptr)
                 : "memory");
}
__device__ __forceinline__ void wait_async0() {
    asm volatile("s_wait_asynccnt 0" ::: "memory");
}

// ---------------- 1) init scratch state ----------------
__global__ void moe_init(int* __restrict__ perm, int* __restrict__ counts,
                         float* __restrict__ psum, int* __restrict__ cursor) {
    int i = blockIdx.x * blockDim.x + threadIdx.x;
    if (i < PADT) perm[i] = -1;
    if (i < Em) { counts[i] = 0; psum[i] = 0.0f; cursor[i] = 0; }
}

// ---------------- 2) router: logits/softmax/top1 + x->bf16 ----------------
__global__ __launch_bounds__(256)
void moe_router(const float* __restrict__ x, const float* __restrict__ rw,
                const float* __restrict__ rb, unsigned short* __restrict__ x_bf,
                float* __restrict__ max_prob, int* __restrict__ max_idx,
                int* __restrict__ counts, float* __restrict__ psum) {
    int t = blockIdx.x, tid = threadIdx.x;
    __shared__ float red[256 * Em];

    const float* xr = x + (size_t)t * Dm + tid * 4;
    float xa[4] = { xr[0], xr[1], xr[2], xr[3] };

    // fused f32 -> bf16 conversion of x (needed by gemm1)
    unsigned int p0 = (unsigned)f2bf(xa[0]) | ((unsigned)f2bf(xa[1]) << 16);
    unsigned int p1 = (unsigned)f2bf(xa[2]) | ((unsigned)f2bf(xa[3]) << 16);
    *(uint2*)(x_bf + (size_t)t * Dm + tid * 4) = make_uint2(p0, p1);

    float part[Em];
#pragma unroll
    for (int e = 0; e < Em; ++e) part[e] = 0.0f;
    const float* w0 = rw + (size_t)(tid * 4) * Em;   // router_w is [D,E] row-major
#pragma unroll
    for (int i = 0; i < 4; ++i) {
        const float* wr = w0 + i * Em;
#pragma unroll
        for (int e = 0; e < Em; ++e) part[e] += xa[i] * wr[e];
    }
#pragma unroll
    for (int e = 0; e < Em; ++e) red[tid * Em + e] = part[e];
    __syncthreads();
    for (int s = 128; s >= 1; s >>= 1) {
        if (tid < s) {
#pragma unroll
            for (int e = 0; e < Em; ++e) red[tid * Em + e] += red[(tid + s) * Em + e];
        }
        __syncthreads();
    }
    if (tid == 0) {
        float lg[Em], mx = -1e30f;
#pragma unroll
        for (int e = 0; e < Em; ++e) { lg[e] = red[e] + rb[e]; mx = fmaxf(mx, lg[e]); }
        float sum = 0.0f, p[Em];
#pragma unroll
        for (int e = 0; e < Em; ++e) { p[e] = __expf(lg[e] - mx); sum += p[e]; }
        float inv = 1.0f / sum, best = -1.0f; int bi = 0;
#pragma unroll
        for (int e = 0; e < Em; ++e) {
            p[e] *= inv;
            atomicAdd(&psum[e], p[e]);
            if (p[e] > best) { best = p[e]; bi = e; }   // first-max tie-break, like argmax
        }
        max_prob[t] = best;
        max_idx[t]  = bi;
        atomicAdd(&counts[bi], 1);
    }
}

// ---------------- 3) scan: padded offsets, tile map, balance loss ----------------
__global__ void moe_scan(const int* __restrict__ counts, const float* __restrict__ psum,
                         int* __restrict__ cursor, int* __restrict__ tileE,
                         int* __restrict__ tileR, float* __restrict__ loss_out) {
    int row = 0, rt = 0;
    float loss = 0.0f;
    for (int e = 0; e < Em; ++e) {
        int c = counts[e];
        cursor[e] = row;                       // scatter write base for expert e
        int nt = (c + BMt - 1) / BMt;
        for (int i = 0; i < nt; ++i) { tileE[rt] = e; tileR[rt] = row + i * BMt; ++rt; }
        row += nt * BMt;
        loss += (psum[e] / (float)T_TOK) * ((float)c / (float)T_TOK);
    }
    for (; rt < ROW_TILES; ++rt) { tileE[rt] = -1; tileR[rt] = 0; }
    loss_out[0] = loss * (float)Em;
}

// ---------------- 4) scatter tokens into expert groups ----------------
__global__ void moe_scatter(const int* __restrict__ max_idx, int* __restrict__ cursor,
                            int* __restrict__ perm) {
    int t = blockIdx.x * blockDim.x + threadIdx.x;
    if (t >= T_TOK) return;
    int e = max_idx[t];
    int pos = atomicAdd(&cursor[e], 1);
    perm[pos] = t;
}

// ---------------- 5) gemm1: h = gelu(Xg @ W1[e] + b1[e]) -> bf16 ----------------
__global__ __launch_bounds__(256)
void moe_gemm1(const unsigned short* __restrict__ x_bf, const float* __restrict__ w1,
               const float* __restrict__ b1, const int* __restrict__ perm,
               const int* __restrict__ tileE, const int* __restrict__ tileR,
               unsigned short* __restrict__ h_bf) {
    int rt = blockIdx.x;
    int e = tileE[rt];
    if (e < 0) return;
    int row0 = tileR[rt];
    int nb = blockIdx.y * BNt;

    __shared__ unsigned short sA[BMt * LDA];
    __shared__ unsigned short sB[BNt * LDB];
    __shared__ int sTok[BMt];

    int tid = threadIdx.x;
    if (tid < BMt) sTok[tid] = perm[row0 + tid];

    int lane = tid & 31, wave = tid >> 5;
    int mOff = (wave & 1) * 32;          // 2 waves along M (64 rows)
    int nOffW = (wave >> 1) * 32;        // 4 waves along N (128 cols)
    int lhalf = lane >> 4, lmod = lane & 15;

    v8f acc00 = {0,0,0,0,0,0,0,0}, acc01 = acc00, acc10 = acc00, acc11 = acc00;

    const float* w1e = w1 + (size_t)e * Dm * Hm;
    int mA = tid >> 2, kA = (tid & 3) * 8;      // A: 8 bf16 per thread
    int kB = tid >> 3, nB = (tid & 7) * 16;     // B: 16 f32->bf16 per thread
    unsigned ldsA_dst = lds_addr(sA + mA * LDA + kA);

    for (int kb = 0; kb < Dm; kb += BKt) {
        __syncthreads();
        // stage A tile [64 x 32]: direct async global->LDS for valid rows,
        // zero-fill padding rows through DS.
        int t = sTok[mA];
        if (t >= 0) {
            async_copy_b128(ldsA_dst, x_bf + (size_t)t * Dm + kb + kA);
        } else {
            *(uint4*)(sA + mA * LDA + kA) = make_uint4(0u, 0u, 0u, 0u);
        }
        // stage B tile [32 x 128], f32->bf16, transposed to N-major for contiguous frags
        const float* src = w1e + (size_t)(kb + kB) * Hm + nb + nB;
#pragma unroll
        for (int j = 0; j < 16; ++j) sB[(nB + j) * LDB + kB] = f2bf(src[j]);
        if (kb + BKt < Dm)
            __builtin_prefetch(w1e + (size_t)(kb + BKt + kB) * Hm + nb + nB, 0, 3);
        wait_async0();
        __syncthreads();

        // A frag (16x32, M=lane%16; K halves per lane-half; pairs packed per VGPR)
        Frag a0, a1, b0, b1;
        int ar = mOff + lmod, akb = lhalf * 16;
        a0.q[0] = *(const uint4*)(sA + ar * LDA + akb);
        a0.q[1] = *(const uint4*)(sA + ar * LDA + akb + 8);
        a1.q[0] = *(const uint4*)(sA + (ar + 16) * LDA + akb);
        a1.q[1] = *(const uint4*)(sA + (ar + 16) * LDA + akb + 8);
        // B frag (32x16, N=lane%16, K=(lane/16)*16 + 0..15 contiguous in N-major LDS)
        int bc = nOffW + lmod, bkb = lhalf * 16;
        b0.q[0] = *(const uint4*)(sB + bc * LDB + bkb);
        b0.q[1] = *(const uint4*)(sB + bc * LDB + bkb + 8);
        b1.q[0] = *(const uint4*)(sB + (bc + 16) * LDB + bkb);
        b1.q[1] = *(const uint4*)(sB + (bc + 16) * LDB + bkb + 8);

        acc00 = __builtin_amdgcn_wmma_f32_16x16x32_bf16(false, a0.v, false, b0.v, (short)0, acc00, false, false);
        acc01 = __builtin_amdgcn_wmma_f32_16x16x32_bf16(false, a0.v, false, b1.v, (short)0, acc01, false, false);
        acc10 = __builtin_amdgcn_wmma_f32_16x16x32_bf16(false, a1.v, false, b0.v, (short)0, acc10, false, false);
        acc11 = __builtin_amdgcn_wmma_f32_16x16x32_bf16(false, a1.v, false, b1.v, (short)0, acc11, false, false);
    }

    // epilogue: bias + gelu -> bf16 h (skip padding rows)
    v8f acc[2][2] = { { acc00, acc01 }, { acc10, acc11 } };
#pragma unroll
    for (int i = 0; i < 2; ++i)
#pragma unroll
        for (int j = 0; j < 2; ++j)
#pragma unroll
            for (int r = 0; r < 8; ++r) {
                int m = mOff + i * 16 + lhalf * 8 + r;
                int t = sTok[m];
                if (t < 0) continue;
                int col = nb + nOffW + j * 16 + lmod;
                float v = acc[i][j][r] + b1[e * Hm + col];
                h_bf[(size_t)(row0 + m) * Hm + col] = f2bf(gelu_tanh(v));
            }
}

// ---------------- 6) gemm2: out[t] = (h @ W2[e] + b2[e]) * prob[t] ----------------
__global__ __launch_bounds__(256)
void moe_gemm2(const unsigned short* __restrict__ h_bf, const float* __restrict__ w2,
               const float* __restrict__ b2, const int* __restrict__ perm,
               const int* __restrict__ tileE, const int* __restrict__ tileR,
               const float* __restrict__ max_prob, float* __restrict__ out) {
    int rt = blockIdx.x;
    int e = tileE[rt];
    if (e < 0) return;
    int row0 = tileR[rt];
    int nb = blockIdx.y * BNt;

    __shared__ unsigned short sA[BMt * LDA];
    __shared__ unsigned short sB[BNt * LDB];
    __shared__ int sTok[BMt];

    int tid = threadIdx.x;
    if (tid < BMt) sTok[tid] = perm[row0 + tid];

    int lane = tid & 31, wave = tid >> 5;
    int mOff = (wave & 1) * 32;
    int nOffW = (wave >> 1) * 32;
    int lhalf = lane >> 4, lmod = lane & 15;

    v8f acc00 = {0,0,0,0,0,0,0,0}, acc01 = acc00, acc10 = acc00, acc11 = acc00;

    const float* w2e = w2 + (size_t)e * Hm * Dm;
    int mA = tid >> 2, kA = (tid & 3) * 8;
    int kB = tid >> 3, nB = (tid & 7) * 16;
    unsigned ldsA_dst = lds_addr(sA + mA * LDA + kA);
    const unsigned short* aRow = h_bf + (size_t)(row0 + mA) * Hm + kA;

    for (int kb = 0; kb < Hm; kb += BKt) {
        __syncthreads();
        // A rows are grouped-contiguous in h_bf: pure async block copy to LDS
        // (padding rows carry garbage but their stores are skipped in epilogue)
        async_copy_b128(ldsA_dst, aRow + kb);
        const float* src = w2e + (size_t)(kb + kB) * Dm + nb + nB;
#pragma unroll
        for (int j = 0; j < 16; ++j) sB[(nB + j) * LDB + kB] = f2bf(src[j]);
        if (kb + BKt < Hm)
            __builtin_prefetch(w2e + (size_t)(kb + BKt + kB) * Dm + nb + nB, 0, 3);
        wait_async0();
        __syncthreads();

        Frag a0, a1, b0, b1;
        int ar = mOff + lmod, akb = lhalf * 16;
        a0.q[0] = *(const uint4*)(sA + ar * LDA + akb);
        a0.q[1] = *(const uint4*)(sA + ar * LDA + akb + 8);
        a1.q[0] = *(const uint4*)(sA + (ar + 16) * LDA + akb);
        a1.q[1] = *(const uint4*)(sA + (ar + 16) * LDA + akb + 8);
        int bc = nOffW + lmod, bkb = lhalf * 16;
        b0.q[0] = *(const uint4*)(sB + bc * LDB + bkb);
        b0.q[1] = *(const uint4*)(sB + bc * LDB + bkb + 8);
        b1.q[0] = *(const uint4*)(sB + (bc + 16) * LDB + bkb);
        b1.q[1] = *(const uint4*)(sB + (bc + 16) * LDB + bkb + 8);

        acc00 = __builtin_amdgcn_wmma_f32_16x16x32_bf16(false, a0.v, false, b0.v, (short)0, acc00, false, false);
        acc01 = __builtin_amdgcn_wmma_f32_16x16x32_bf16(false, a0.v, false, b1.v, (short)0, acc01, false, false);
        acc10 = __builtin_amdgcn_wmma_f32_16x16x32_bf16(false, a1.v, false, b0.v, (short)0, acc10, false, false);
        acc11 = __builtin_amdgcn_wmma_f32_16x16x32_bf16(false, a1.v, false, b1.v, (short)0, acc11, false, false);
    }

    v8f acc[2][2] = { { acc00, acc01 }, { acc10, acc11 } };
#pragma unroll
    for (int i = 0; i < 2; ++i)
#pragma unroll
        for (int j = 0; j < 2; ++j)
#pragma unroll
            for (int r = 0; r < 8; ++r) {
                int m = mOff + i * 16 + lhalf * 8 + r;
                int t = sTok[m];
                if (t < 0) continue;
                int col = nb + nOffW + j * 16 + lmod;
                float v = acc[i][j][r] + b2[e * Dm + col];
                out[(size_t)t * Dm + col] = v * max_prob[t];
            }
}

// ---------------- host launcher ----------------
extern "C" void kernel_launch(void* const* d_in, const int* in_sizes, int n_in,
                              void* d_out, int out_size, void* d_ws, size_t ws_size,
                              hipStream_t stream) {
    (void)in_sizes; (void)n_in; (void)out_size; (void)ws_size;
    const float* x  = (const float*)d_in[0];
    const float* rw = (const float*)d_in[1];
    const float* rb = (const float*)d_in[2];
    const float* w1 = (const float*)d_in[3];
    const float* b1 = (const float*)d_in[4];
    const float* w2 = (const float*)d_in[5];
    const float* b2 = (const float*)d_in[6];
    float* out = (float*)d_out;

    char* p = (char*)d_ws;
    auto alloc = [&](size_t bytes) -> char* {
        char* r = p; p += (bytes + 255) & ~(size_t)255; return r;
    };
    unsigned short* x_bf  = (unsigned short*)alloc((size_t)T_TOK * Dm * 2);
    unsigned short* h_bf  = (unsigned short*)alloc((size_t)PADT * Hm * 2);
    int*   perm   = (int*)  alloc((size_t)PADT * 4);
    float* mprob  = (float*)alloc((size_t)T_TOK * 4);
    int*   midx   = (int*)  alloc((size_t)T_TOK * 4);
    int*   counts = (int*)  alloc(Em * 4);
    float* psum   = (float*)alloc(Em * 4);
    int*   cursor = (int*)  alloc(Em * 4);
    int*   tileE  = (int*)  alloc(ROW_TILES * 4);
    int*   tileR  = (int*)  alloc(ROW_TILES * 4);

    moe_init<<<(PADT + 255) / 256, 256, 0, stream>>>(perm, counts, psum, cursor);
    moe_router<<<T_TOK, 256, 0, stream>>>(x, rw, rb, x_bf, mprob, midx, counts, psum);
    moe_scan<<<1, 1, 0, stream>>>(counts, psum, cursor, tileE, tileR,
                                  out + (size_t)T_TOK * Dm);
    moe_scatter<<<T_TOK / 256, 256, 0, stream>>>(midx, cursor, perm);
    moe_gemm1<<<dim3(ROW_TILES, Hm / BNt), 256, 0, stream>>>(x_bf, w1, b1, perm,
                                                             tileE, tileR, h_bf);
    moe_gemm2<<<dim3(ROW_TILES, Dm / BNt), 256, 0, stream>>>(h_bf, w2, b2, perm,
                                                             tileE, tileR, mprob, out);
}
